// AttentionLayer_26912265076751
// MI455X (gfx1250) — matrix operand order repaired and checked by reading, not compiled, not run on hardware
//
#include <hip/hip_runtime.h>

// ---------------------------------------------------------------------------
// CDNA5 (gfx1250) attention layer: bf16 WMMA (16x16x32) with fp32 accumulate.
// K-loop unrolled x2 with ping-pong fragment buffers: no register-rotation
// movs, loads for step k+1 in flight while WMMAs consume step k.
// ---------------------------------------------------------------------------

#define DM 1024          // model dim
#define SEQ 2048         // sequence length
#define NB 4             // batch
#define ROWS (NB*SEQ)    // 8192 flattened (b,s) rows

typedef __attribute__((ext_vector_type(16))) __bf16 v16bf;
typedef __attribute__((ext_vector_type(8)))  float  v8f;

union ABFrag { v16bf v; uint4 u[2]; };
union Pack8  { uint4 u; __bf16 h[8]; };

// A-fragment (16x32, bf16): lane holds row M = lane%16.
// element e -> K = k0 + 16*(lane/16) + e
__device__ __forceinline__ v16bf load_a_frag(const __bf16* row_base, int k0, int half) {
    ABFrag f;
    f.u[0] = *(const uint4*)(row_base + k0 + 16 * half);
    f.u[1] = *(const uint4*)(row_base + k0 + 16 * half + 8);
    return f.v;
}

// B-fragment (32x16, bf16): lane holds column N = lane%16.
// element e -> K = k0 + 16*(lane/16) + e  (one contiguous 32B chunk)
__device__ __forceinline__ v16bf load_b_frag(const __bf16* col_base, int k0, int half) {
    ABFrag f;
    const __bf16* p = col_base + k0 + 16 * half;
    f.u[0] = *(const uint4*)(p);
    f.u[1] = *(const uint4*)(p + 8);
    return f.v;
}

__device__ __forceinline__ v8f wmma_bf16(v16bf a, v16bf b, v8f c) {
    return __builtin_amdgcn_wmma_f32_16x16x32_bf16(false, a, false, b, (short)0, c, false, false);
}

// ---------------------------------------------------------------------------
// Ping-pong pipelined 2(M) x 4(N) x K core. acc[0..3]=M-tile0, acc[4..7]=M-tile1.
// K must be a multiple of 64.
// ---------------------------------------------------------------------------
template<int K>
__device__ __forceinline__ void mm_2x4(const __bf16* a0, const __bf16* a1,
                                       const __bf16* b0, const __bf16* b1,
                                       const __bf16* b2, const __bf16* b3,
                                       int half, v8f acc[8]) {
    const __bf16* bp[4] = {b0, b1, b2, b3};
    v16bf A0a, A1a, Ba[4];   // even k-step buffers
    v16bf A0b, A1b, Bb[4];   // odd  k-step buffers

    A0a = load_a_frag(a0, 0, half);
    A1a = load_a_frag(a1, 0, half);
#pragma unroll
    for (int j = 0; j < 4; ++j) Ba[j] = load_b_frag(bp[j], 0, half);

#pragma unroll 1
    for (int i = 0; i < K / 64 - 1; ++i) {
        const int kA = i * 64;
        // stage odd buffer @ kA+32
        A0b = load_a_frag(a0, kA + 32, half);
        A1b = load_a_frag(a1, kA + 32, half);
#pragma unroll
        for (int j = 0; j < 4; ++j) Bb[j] = load_b_frag(bp[j], kA + 32, half);
        // consume even buffer
#pragma unroll
        for (int j = 0; j < 4; ++j) {
            acc[j]     = wmma_bf16(A0a, Ba[j], acc[j]);
            acc[4 + j] = wmma_bf16(A1a, Ba[j], acc[4 + j]);
        }
        // stage even buffer @ kA+64
        A0a = load_a_frag(a0, kA + 64, half);
        A1a = load_a_frag(a1, kA + 64, half);
#pragma unroll
        for (int j = 0; j < 4; ++j) Ba[j] = load_b_frag(bp[j], kA + 64, half);
        // consume odd buffer
#pragma unroll
        for (int j = 0; j < 4; ++j) {
            acc[j]     = wmma_bf16(A0b, Bb[j], acc[j]);
            acc[4 + j] = wmma_bf16(A1b, Bb[j], acc[4 + j]);
        }
    }
    // tail: last 64 of K
    A0b = load_a_frag(a0, K - 32, half);
    A1b = load_a_frag(a1, K - 32, half);
#pragma unroll
    for (int j = 0; j < 4; ++j) Bb[j] = load_b_frag(bp[j], K - 32, half);
#pragma unroll
    for (int j = 0; j < 4; ++j) {
        acc[j]     = wmma_bf16(A0a, Ba[j], acc[j]);
        acc[4 + j] = wmma_bf16(A1a, Ba[j], acc[4 + j]);
    }
#pragma unroll
    for (int j = 0; j < 4; ++j) {
        acc[j]     = wmma_bf16(A0b, Bb[j], acc[j]);
        acc[4 + j] = wmma_bf16(A1b, Bb[j], acc[4 + j]);
    }
}

// ---------------------------------------------------------------------------
// Ping-pong pipelined 1(M) x 4(N) x K core (A may live in LDS or global).
// ---------------------------------------------------------------------------
template<int K>
__device__ __forceinline__ void mm_1x4(const __bf16* a0,
                                       const __bf16* b0, const __bf16* b1,
                                       const __bf16* b2, const __bf16* b3,
                                       int half, v8f acc[4]) {
    const __bf16* bp[4] = {b0, b1, b2, b3};
    v16bf Aa, Ba[4];
    v16bf Ab, Bb[4];

    Aa = load_a_frag(a0, 0, half);
#pragma unroll
    for (int j = 0; j < 4; ++j) Ba[j] = load_b_frag(bp[j], 0, half);

#pragma unroll 1
    for (int i = 0; i < K / 64 - 1; ++i) {
        const int kA = i * 64;
        Ab = load_a_frag(a0, kA + 32, half);
#pragma unroll
        for (int j = 0; j < 4; ++j) Bb[j] = load_b_frag(bp[j], kA + 32, half);
#pragma unroll
        for (int j = 0; j < 4; ++j) acc[j] = wmma_bf16(Aa, Ba[j], acc[j]);
        Aa = load_a_frag(a0, kA + 64, half);
#pragma unroll
        for (int j = 0; j < 4; ++j) Ba[j] = load_b_frag(bp[j], kA + 64, half);
#pragma unroll
        for (int j = 0; j < 4; ++j) acc[j] = wmma_bf16(Ab, Bb[j], acc[j]);
    }
    Ab = load_a_frag(a0, K - 32, half);
#pragma unroll
    for (int j = 0; j < 4; ++j) Bb[j] = load_b_frag(bp[j], K - 32, half);
#pragma unroll
    for (int j = 0; j < 4; ++j) acc[j] = wmma_bf16(Aa, Ba[j], acc[j]);
#pragma unroll
    for (int j = 0; j < 4; ++j) acc[j] = wmma_bf16(Ab, Bb[j], acc[j]);
}

// ---------------------------------------------------------------------------
// fp32 -> bf16 conversion
// ---------------------------------------------------------------------------
__global__ void cvt_f32_bf16(const float* __restrict__ src, __bf16* __restrict__ dst, int n) {
    int i = blockIdx.x * blockDim.x + threadIdx.x;
    if (i < n) dst[i] = (__bf16)src[i];
}

// ---------------------------------------------------------------------------
// QKV GEMM: [8192,1024] x [3072,1024]^T + bias -> q,k row-major, v transposed.
// Block = 4 waves; each wave computes a 32x64 output tile (2x4 WMMA tiles).
// ---------------------------------------------------------------------------
__global__ __launch_bounds__(128)
void qkv_gemm(const __bf16* __restrict__ X,     // [ROWS, DM]
              const __bf16* __restrict__ W,     // [3*DM, DM]
              const float*  __restrict__ bias,  // [3*DM]
              __bf16* __restrict__ Q,           // [ROWS, DM]
              __bf16* __restrict__ Kb,          // [ROWS, DM]
              __bf16* __restrict__ VT)          // [NB][DM][SEQ]
{
    const int lane = threadIdx.x & 31;
    const int wave = threadIdx.x >> 5;
    const int half = lane >> 4;
    const int lm   = lane & 15;
    const int mt0  = blockIdx.y * 8 + wave * 2;  // first of 2 M tiles
    const int nb   = blockIdx.x * 64;            // 64-col slab base

    const __bf16* a0 = X + (long)(mt0 * 16 + lm) * DM;
    const __bf16* a1 = X + (long)(mt0 * 16 + 16 + lm) * DM;
    const __bf16* b0 = W + (long)(nb + 0 * 16 + lm) * DM;
    const __bf16* b1 = W + (long)(nb + 1 * 16 + lm) * DM;
    const __bf16* b2 = W + (long)(nb + 2 * 16 + lm) * DM;
    const __bf16* b3 = W + (long)(nb + 3 * 16 + lm) * DM;

    v8f acc[8];
#pragma unroll
    for (int i = 0; i < 8; ++i) acc[i] = v8f{};
    mm_2x4<DM>(a0, a1, b0, b1, b2, b3, half, acc);

#pragma unroll
    for (int j = 0; j < 4; ++j) {
        const int col = nb + j * 16 + lm;     // C-layout: N = lane%16
        const float bv = bias[col];
#pragma unroll
        for (int t = 0; t < 2; ++t) {
#pragma unroll
            for (int r = 0; r < 8; ++r) {
                const int orow = (mt0 + t) * 16 + r + 8 * half;
                const float val = acc[t * 4 + j][r] + bv;
                if (col < DM) {
                    Q[(long)orow * DM + col] = (__bf16)val;
                } else if (col < 2 * DM) {
                    Kb[(long)orow * DM + (col - DM)] = (__bf16)val;
                } else {
                    const int b = orow >> 11;          // / SEQ
                    const int s = orow & (SEQ - 1);
                    VT[((long)b * DM + (col - 2 * DM)) * SEQ + s] = (__bf16)val;
                }
            }
        }
    }
}

// ---------------------------------------------------------------------------
// Fused attention per (batch, 16-query tile). One wave per block.
// scores -> LDS (bf16) -> softmax -> P@V via WMMA with LDS A-fragments.
// ---------------------------------------------------------------------------
__global__ __launch_bounds__(32)
void attn_kernel(const __bf16* __restrict__ Q,
                 const __bf16* __restrict__ Kb,
                 const __bf16* __restrict__ VT,   // [NB][DM][SEQ]
                 __bf16* __restrict__ O)          // [ROWS, DM]
{
    __shared__ __align__(16) __bf16 sP[16 * SEQ]; // 64 KB: scores, then exp-weights
    __shared__ float sInv[16];

    const int lane = threadIdx.x & 31;
    const int half = lane >> 4;
    const int lm   = lane & 15;
    const int b    = blockIdx.x >> 7;       // SEQ/16 = 128 q-tiles per batch
    const int qt   = blockIdx.x & 127;
    const float scale = 0.03125f;           // 1/sqrt(1024)

    const __bf16* qrow = Q + (long)(b * SEQ + qt * 16 + lm) * DM;

    // ---- Pass 1: scores = q_tile @ k^T (4 key tiles per group), into LDS ----
    for (int ktg = 0; ktg < SEQ / 64; ++ktg) {
        const __bf16* k0p = Kb + (long)(b * SEQ + ktg * 64 + 0 * 16 + lm) * DM;
        const __bf16* k1p = Kb + (long)(b * SEQ + ktg * 64 + 1 * 16 + lm) * DM;
        const __bf16* k2p = Kb + (long)(b * SEQ + ktg * 64 + 2 * 16 + lm) * DM;
        const __bf16* k3p = Kb + (long)(b * SEQ + ktg * 64 + 3 * 16 + lm) * DM;
        v8f acc[4];
#pragma unroll
        for (int i = 0; i < 4; ++i) acc[i] = v8f{};
        mm_1x4<DM>(qrow, k0p, k1p, k2p, k3p, half, acc);
#pragma unroll
        for (int j = 0; j < 4; ++j) {
#pragma unroll
            for (int r = 0; r < 8; ++r) {
                const int m = r + 8 * half;            // query row within tile
                const int n = ktg * 64 + j * 16 + lm;  // key index
                sP[m * SEQ + n] = (__bf16)(acc[j][r] * scale);
            }
        }
    }
    __syncthreads();

    // ---- Pass 2: row softmax (lanes 0..15, one row each), 8-wide vectors ----
    if (lane < 16) {
        const int base = lane * SEQ;
        float mx = -3.0e38f;
        for (int j = 0; j < SEQ; j += 8) {
            Pack8 u; u.u = *(const uint4*)&sP[base + j];
#pragma unroll
            for (int t = 0; t < 8; ++t) mx = fmaxf(mx, (float)u.h[t]);
        }
        float sum = 0.0f;
        for (int j = 0; j < SEQ; j += 8) {
            Pack8 u; u.u = *(const uint4*)&sP[base + j];
            Pack8 o;
#pragma unroll
            for (int t = 0; t < 8; ++t) {
                float e = __expf((float)u.h[t] - mx);
                sum += e;
                o.h[t] = (__bf16)e;                    // unnormalized weight
            }
            *(uint4*)&sP[base + j] = o.u;
        }
        sInv[lane] = 1.0f / sum;
    }
    __syncthreads();

    // ---- Pass 3: O_tile = (P @ V) * (1/rowsum), 4 V-column tiles per group ----
    const __bf16* prow = &sP[lm * SEQ];
    for (int ntg = 0; ntg < DM / 64; ++ntg) {
        const __bf16* v0p = VT + ((long)b * DM + ntg * 64 + 0 * 16 + lm) * SEQ;
        const __bf16* v1p = VT + ((long)b * DM + ntg * 64 + 1 * 16 + lm) * SEQ;
        const __bf16* v2p = VT + ((long)b * DM + ntg * 64 + 2 * 16 + lm) * SEQ;
        const __bf16* v3p = VT + ((long)b * DM + ntg * 64 + 3 * 16 + lm) * SEQ;
        v8f acc[4];
#pragma unroll
        for (int i = 0; i < 4; ++i) acc[i] = v8f{};
        mm_1x4<SEQ>(prow, v0p, v1p, v2p, v3p, half, acc);
#pragma unroll
        for (int j = 0; j < 4; ++j) {
#pragma unroll
            for (int r = 0; r < 8; ++r) {
                const int m = r + 8 * half;
                const int orow = b * SEQ + qt * 16 + m;
                const int col = ntg * 64 + j * 16 + lm;
                O[(long)orow * DM + col] = (__bf16)(acc[j][r] * sInv[m]);
            }
        }
    }
}

// ---------------------------------------------------------------------------
// Output projection: [8192,1024] x [1024,1024]^T + bias -> fp32 d_out
// ---------------------------------------------------------------------------
__global__ __launch_bounds__(128)
void proj_gemm(const __bf16* __restrict__ A,     // [ROWS, DM]
               const __bf16* __restrict__ W,     // [DM, DM]
               const float*  __restrict__ bias,  // [DM]
               float* __restrict__ Out)          // [ROWS, DM]
{
    const int lane = threadIdx.x & 31;
    const int wave = threadIdx.x >> 5;
    const int half = lane >> 4;
    const int lm   = lane & 15;
    const int mt0  = blockIdx.y * 8 + wave * 2;
    const int nb   = blockIdx.x * 64;

    const __bf16* a0 = A + (long)(mt0 * 16 + lm) * DM;
    const __bf16* a1 = A + (long)(mt0 * 16 + 16 + lm) * DM;
    const __bf16* b0 = W + (long)(nb + 0 * 16 + lm) * DM;
    const __bf16* b1 = W + (long)(nb + 1 * 16 + lm) * DM;
    const __bf16* b2 = W + (long)(nb + 2 * 16 + lm) * DM;
    const __bf16* b3 = W + (long)(nb + 3 * 16 + lm) * DM;

    v8f acc[8];
#pragma unroll
    for (int i = 0; i < 8; ++i) acc[i] = v8f{};
    mm_2x4<DM>(a0, a1, b0, b1, b2, b3, half, acc);

#pragma unroll
    for (int j = 0; j < 4; ++j) {
        const int col = nb + j * 16 + lm;
        const float bv = bias[col];
#pragma unroll
        for (int t = 0; t < 2; ++t) {
#pragma unroll
            for (int r = 0; r < 8; ++r) {
                const int orow = (mt0 + t) * 16 + r + 8 * half;
                Out[(long)orow * DM + col] = acc[t * 4 + j][r] + bv;
            }
        }
    }
}

// ---------------------------------------------------------------------------
// Host launcher
// ---------------------------------------------------------------------------
extern "C" void kernel_launch(void* const* d_in, const int* in_sizes, int n_in,
                              void* d_out, int out_size, void* d_ws, size_t ws_size,
                              hipStream_t stream) {
    const float* x     = (const float*)d_in[0];   // [4,2048,1024]
    const float* Wqkv  = (const float*)d_in[1];   // [3072,1024]
    const float* bqkv  = (const float*)d_in[2];   // [3072]
    const float* Wproj = (const float*)d_in[3];   // [1024,1024]
    const float* bproj = (const float*)d_in[4];   // [1024]
    float* out = (float*)d_out;                   // [4,2048,1024] fp32

    // Workspace layout (bf16), ~88 MB total
    char* ws = (char*)d_ws;
    const size_t N_X  = (size_t)ROWS * DM;        // 8388608
    const size_t N_WQ = (size_t)3 * DM * DM;      // 3145728
    const size_t N_WP = (size_t)DM * DM;          // 1048576
    size_t off = 0;
    __bf16* xb    = (__bf16*)(ws + off); off += N_X  * 2;
    __bf16* wqkvb = (__bf16*)(ws + off); off += N_WQ * 2;
    __bf16* wprojb= (__bf16*)(ws + off); off += N_WP * 2;
    __bf16* qb    = (__bf16*)(ws + off); off += N_X  * 2;
    __bf16* kb    = (__bf16*)(ws + off); off += N_X  * 2;
    __bf16* vT    = (__bf16*)(ws + off); off += N_X  * 2;
    __bf16* ob    = (__bf16*)(ws + off); off += N_X  * 2;

    // 1) fp32 -> bf16 conversions
    cvt_f32_bf16<<<dim3((unsigned)((N_X  + 255) / 256)), dim3(256), 0, stream>>>(x,     xb,     (int)N_X);
    cvt_f32_bf16<<<dim3((unsigned)((N_WQ + 255) / 256)), dim3(256), 0, stream>>>(Wqkv,  wqkvb,  (int)N_WQ);
    cvt_f32_bf16<<<dim3((unsigned)((N_WP + 255) / 256)), dim3(256), 0, stream>>>(Wproj, wprojb, (int)N_WP);

    // 2) QKV projection (q,k row-major; v transposed for P@V B-fragments)
    qkv_gemm<<<dim3(3 * DM / 64, ROWS / 128), dim3(128), 0, stream>>>(
        xb, wqkvb, bqkv, qb, kb, vT);

    // 3) fused attention: one wave per (batch, 16-query tile)
    attn_kernel<<<dim3(NB * SEQ / 16), dim3(32), 0, stream>>>(qb, kb, vT, ob);

    // 4) output projection + bias -> fp32
    proj_gemm<<<dim3(DM / 64, ROWS / 128), dim3(128), 0, stream>>>(
        ob, wprojb, bproj, out);
}
